// GraphSAGE_81320910783036
// MI455X (gfx1250) — matrix operand order repaired and checked by reading, not compiled, not run on hardware
//
#include <hip/hip_runtime.h>

// ---------------------------------------------------------------------------
// GraphSAGE on gfx1250 (MI455X, wave32, WMMA).
// h1 = relu(concat(x, max_neigh(x)) @ W1 + b1)          for all 100k nodes
// out = (concat(h1,max_neigh(h1))[batch] @ W2 + b2) @ Wout + bout   (8192 rows)
// GEMMs via v_wmma_f32_16x16x32_f16 (f16 inputs, f32 accumulate).
// ---------------------------------------------------------------------------

typedef __attribute__((ext_vector_type(16))) _Float16 v16h;
typedef __attribute__((ext_vector_type(8)))  _Float16 v8h;
typedef __attribute__((ext_vector_type(4)))  _Float16 h4;
typedef __attribute__((ext_vector_type(8)))  float    v8f;

#define N_NODES 100000
#define SAMPLES 16
#define D       128
#define NB      8192

#define FSTR 264   // feats row stride (halves): 256 + 8 pad, keeps 16B align, spreads banks
#define HSTR 136   // h2 row stride (halves): 128 + 8 pad

static __device__ __forceinline__ v8f wmma_f16(v16h a, v16h b, v8f c) {
    return __builtin_amdgcn_wmma_f32_16x16x32_f16(false, a, false, b, (short)0, c,
                                                  false, false);
}

// --- K1: transpose + f16-convert the weight matrices into workspace ---------
__global__ __launch_bounds__(256) void prep_weights(
    const float* __restrict__ W1, const float* __restrict__ W2,
    const float* __restrict__ Wout,
    _Float16* __restrict__ Wt1, _Float16* __restrict__ Wt2,
    _Float16* __restrict__ WoT)
{
    int i = blockIdx.x * 256 + threadIdx.x;
    if (i < 32768) {                       // W1 [256][128] -> Wt1 [n][k]
        int k = i >> 7, n = i & 127;
        Wt1[n * 256 + k] = (_Float16)W1[i];
    } else if (i < 65536) {                // W2 [256][128] -> Wt2 [n][k]
        int j = i - 32768;
        int k = j >> 7, n = j & 127;
        Wt2[n * 256 + k] = (_Float16)W2[j];
    } else if (i < 73728) {                // Wout [128][64] -> WoT [n][k]
        int j = i - 65536;
        int k = j >> 6, n = j & 63;
        WoT[n * 128 + k] = (_Float16)Wout[j];
    }
}

// --- K2: layer 1 for all nodes; one wave owns a 16-node tile ----------------
__global__ __launch_bounds__(128) void layer1_kernel(
    const float* __restrict__ x, const int* __restrict__ nidx,
    const _Float16* __restrict__ Wt1, const float* __restrict__ b1,
    _Float16* __restrict__ h1)
{
    __shared__ __align__(16) _Float16 featsAll[4 * 16 * FSTR];

    const int tid  = threadIdx.x;
    const int wave = tid >> 5;
    const int lane = tid & 31;
    const int mrow = lane & 15;
    const int hi   = lane >> 4;            // 0: lanes 0-15, 1: lanes 16-31
    _Float16* feats = featsAll + wave * 16 * FSTR;
    const int tile0 = (blockIdx.x * 4 + wave) * 16;

    // Build feats tile [16][256] = [self(128) | max-agg(128)] in f16.
    for (int m = 0; m < 16; ++m) {
        int node  = tile0 + m;
        int nodec = node < N_NODES ? node : 0;
        const float4* xr = (const float4*)(x + (size_t)nodec * D);
        float4 self = xr[lane];                        // 32 lanes x 4 = 128 dims
        float4 mx = make_float4(-3.4e38f, -3.4e38f, -3.4e38f, -3.4e38f);
        for (int s = 0; s < SAMPLES; ++s) {
            int nb = nidx[(size_t)nodec * SAMPLES + s];
            float4 v = ((const float4*)(x + (size_t)nb * D))[lane];
            mx.x = fmaxf(mx.x, v.x); mx.y = fmaxf(mx.y, v.y);
            mx.z = fmaxf(mx.z, v.z); mx.w = fmaxf(mx.w, v.w);
        }
        _Float16* fs = feats + m * FSTR + lane * 4;
        fs[0] = (_Float16)self.x; fs[1] = (_Float16)self.y;
        fs[2] = (_Float16)self.z; fs[3] = (_Float16)self.w;
        _Float16* fa = feats + m * FSTR + 128 + lane * 4;
        fa[0] = (_Float16)mx.x; fa[1] = (_Float16)mx.y;
        fa[2] = (_Float16)mx.z; fa[3] = (_Float16)mx.w;
    }
    // Wave-local LDS producer/consumer: hardware DScnt waits suffice.

    const v8f vzero = {0.f, 0.f, 0.f, 0.f, 0.f, 0.f, 0.f, 0.f};
    v8f acc[8];
#pragma unroll
    for (int t = 0; t < 8; ++t) acc[t] = vzero;

    // M=16 (nodes), K=256 (feats), N=128: 8 k-steps x 8 n-tiles of WMMA.
#pragma unroll
    for (int kb = 0; kb < 256; kb += 32) {
        union { v16h v; v8h h[2]; } A;     // 16-bit A 16x32 layout (ISA 7.12.2)
        A.h[0] = *(const v8h*)(feats + mrow * FSTR + kb + hi * 8);
        A.h[1] = *(const v8h*)(feats + mrow * FSTR + kb + 16 + hi * 8);
#pragma unroll
        for (int t = 0; t < 8; ++t) {
            const _Float16* wp = Wt1 + (size_t)(t * 16 + mrow) * 256 + kb + hi * 8;
            union { v16h v; v8h h[2]; } B; // 16-bit B 32x16: K = hi*8+0..7, 16+hi*8+0..7 (matches A)
            B.h[0] = *(const v8h*)(wp);
            B.h[1] = *(const v8h*)(wp + 16);
            acc[t] = wmma_f16(A.v, B.v, acc[t]);
        }
    }

    // C layout: VGPR r -> M = r + hi*8, N = mrow + 16*t. Bias + relu -> f16 h1.
#pragma unroll
    for (int t = 0; t < 8; ++t) {
        int n = t * 16 + mrow;
        float bias = b1[n];
#pragma unroll
        for (int r = 0; r < 8; ++r) {
            int node = tile0 + r + hi * 8;
            if (node < N_NODES) {
                float v = acc[t][r] + bias;
                h1[(size_t)node * D + n] = (_Float16)(v > 0.f ? v : 0.f);
            }
        }
    }
}

// --- K3: layer 2 + output projection for the 8192 batch rows ----------------
__global__ __launch_bounds__(128) void layer2_kernel(
    const _Float16* __restrict__ h1, const int* __restrict__ nidx,
    const int* __restrict__ batch,
    const _Float16* __restrict__ Wt2, const float* __restrict__ b2,
    const _Float16* __restrict__ WoT, const float* __restrict__ bout,
    float* __restrict__ out)
{
    __shared__ __align__(16) _Float16 featsAll[4 * 16 * FSTR];
    __shared__ __align__(16) _Float16 h2All[4 * 16 * HSTR];

    const int tid  = threadIdx.x;
    const int wave = tid >> 5;
    const int lane = tid & 31;
    const int mrow = lane & 15;
    const int hi   = lane >> 4;
    _Float16* feats = featsAll + wave * 16 * FSTR;
    _Float16* h2    = h2All + wave * 16 * HSTR;
    const int tile0 = (blockIdx.x * 4 + wave) * 16;   // batch row base (8192 % 64 == 0)

    // feats2 tile from h1 (f16) gathers; h1 >= 0 so f32 max of f16 values is exact.
    for (int m = 0; m < 16; ++m) {
        int node = batch[tile0 + m];
        h4 self = ((const h4*)(h1 + (size_t)node * D))[lane];
        float4 mx = make_float4(-3.4e38f, -3.4e38f, -3.4e38f, -3.4e38f);
        for (int s = 0; s < SAMPLES; ++s) {
            int nb = nidx[(size_t)node * SAMPLES + s];
            h4 v = ((const h4*)(h1 + (size_t)nb * D))[lane];
            mx.x = fmaxf(mx.x, (float)v[0]); mx.y = fmaxf(mx.y, (float)v[1]);
            mx.z = fmaxf(mx.z, (float)v[2]); mx.w = fmaxf(mx.w, (float)v[3]);
        }
        _Float16* fs = feats + m * FSTR + lane * 4;
        fs[0] = self[0]; fs[1] = self[1]; fs[2] = self[2]; fs[3] = self[3];
        _Float16* fa = feats + m * FSTR + 128 + lane * 4;
        fa[0] = (_Float16)mx.x; fa[1] = (_Float16)mx.y;
        fa[2] = (_Float16)mx.z; fa[3] = (_Float16)mx.w;
    }

    const v8f vzero = {0.f, 0.f, 0.f, 0.f, 0.f, 0.f, 0.f, 0.f};
    v8f acc[8];
#pragma unroll
    for (int t = 0; t < 8; ++t) acc[t] = vzero;

    // GEMM1: h2 = feats2 @ W2 (+b2, no relu)
#pragma unroll
    for (int kb = 0; kb < 256; kb += 32) {
        union { v16h v; v8h h[2]; } A;
        A.h[0] = *(const v8h*)(feats + mrow * FSTR + kb + hi * 8);
        A.h[1] = *(const v8h*)(feats + mrow * FSTR + kb + 16 + hi * 8);
#pragma unroll
        for (int t = 0; t < 8; ++t) {
            const _Float16* wp = Wt2 + (size_t)(t * 16 + mrow) * 256 + kb + hi * 8;
            union { v16h v; v8h h[2]; } B;
            B.h[0] = *(const v8h*)(wp);
            B.h[1] = *(const v8h*)(wp + 16);
            acc[t] = wmma_f16(A.v, B.v, acc[t]);
        }
    }
    // Stash h2 (f16, +b2) in LDS as the A operand of the projection GEMM.
#pragma unroll
    for (int t = 0; t < 8; ++t) {
        int n = t * 16 + mrow;
        float bias = b2[n];
#pragma unroll
        for (int r = 0; r < 8; ++r)
            h2[(r + hi * 8) * HSTR + n] = (_Float16)(acc[t][r] + bias);
    }

    // GEMM2: out = h2 @ Wout + bout  (M=16, K=128, N=64)
    v8f acc2[4];
#pragma unroll
    for (int t = 0; t < 4; ++t) acc2[t] = vzero;
#pragma unroll
    for (int kb = 0; kb < 128; kb += 32) {
        union { v16h v; v8h h[2]; } A;
        A.h[0] = *(const v8h*)(h2 + mrow * HSTR + kb + hi * 8);
        A.h[1] = *(const v8h*)(h2 + mrow * HSTR + kb + 16 + hi * 8);
#pragma unroll
        for (int t = 0; t < 4; ++t) {
            const _Float16* wp = WoT + (size_t)(t * 16 + mrow) * 128 + kb + hi * 8;
            union { v16h v; v8h h[2]; } B;
            B.h[0] = *(const v8h*)(wp);
            B.h[1] = *(const v8h*)(wp + 16);
            acc2[t] = wmma_f16(A.v, B.v, acc2[t]);
        }
    }
#pragma unroll
    for (int t = 0; t < 4; ++t) {
        int n = t * 16 + mrow;
        float bo = bout[n];
#pragma unroll
        for (int r = 0; r < 8; ++r) {
            int row = tile0 + r + hi * 8;
            out[(size_t)row * 64 + n] = acc2[t][r] + bo;
        }
    }
}

// ---------------------------------------------------------------------------
extern "C" void kernel_launch(void* const* d_in, const int* in_sizes, int n_in,
                              void* d_out, int out_size, void* d_ws, size_t ws_size,
                              hipStream_t stream) {
    const float* x    = (const float*)d_in[0];
    const int*   nidx = (const int*)d_in[1];
    const int*   batch= (const int*)d_in[2];
    const float* W1   = (const float*)d_in[3];
    const float* b1   = (const float*)d_in[4];
    const float* W2   = (const float*)d_in[5];
    const float* b2   = (const float*)d_in[6];
    const float* Wout = (const float*)d_in[7];
    const float* bout = (const float*)d_in[8];
    float* out = (float*)d_out;

    // Workspace: h1 f16 [100000][128] (25.6 MB) + transposed f16 weights.
    char* ws = (char*)d_ws;
    _Float16* h1  = (_Float16*)ws;
    _Float16* Wt1 = (_Float16*)(ws + (size_t)N_NODES * D * sizeof(_Float16));
    _Float16* Wt2 = Wt1 + 128 * 256;
    _Float16* WoT = Wt2 + 128 * 256;

    prep_weights<<<288, 256, 0, stream>>>(W1, W2, Wout, Wt1, Wt2, WoT);
    layer1_kernel<<<(N_NODES + 63) / 64, 128, 0, stream>>>(x, nidx, Wt1, b1, h1);
    layer2_kernel<<<NB / 64, 128, 0, stream>>>(h1, nidx, batch, Wt2, b2, WoT, bout, out);
}
